// TemporalAttention_15633680957736
// MI455X (gfx1250) — compile-verified, matrix operand order checked by reading, not run on hardware
//
#include <hip/hip_runtime.h>
#include <hip/hip_bf16.h>

typedef __attribute__((ext_vector_type(16))) _Float16 v16h;
typedef __attribute__((ext_vector_type(8)))  float    v8f;
typedef __attribute__((ext_vector_type(4)))  unsigned int u32x4;
typedef __attribute__((ext_vector_type(8)))  int          i32x8;
typedef __attribute__((ext_vector_type(4)))  int          i32x4;
typedef _Float16 half_t;

#if defined(__has_builtin)
# if __has_builtin(__builtin_amdgcn_tensor_load_to_lds) && __has_builtin(__builtin_amdgcn_s_wait_tensorcnt)
#  define USE_TDM 1
# else
#  define USE_TDM 0
# endif
#else
# define USE_TDM 0
#endif

// ---------------------------------------------------------------------------
// Geometry: 256x256 images, fp16 planes with a uniform 3-pixel zero halo so
// both the 3x3 convs (pad 1) and the 7x7 unfold (pad 3) read without branches.
// ---------------------------------------------------------------------------
#define IMG_H 256
#define IMG_W 256
#define HALO  3
#define WPAD  262                 // 256 + 2*3
constexpr long PLANE = 68656;     // 262*262 = 68644 rounded to mult of 16 halves

// plane indices inside the fp16 workspace region
constexpr long P_XA = 0;    // 3 planes
constexpr long P_XB = 3;    // 3
constexpr long P_T1 = 6;    // 48
constexpr long P_T2 = 54;   // 48
constexpr long P_SK = 102;  // 24
constexpr long P_A  = 126;  // 24
constexpr long P_B  = 150;  // 24
constexpr long P_Q  = 174;  // 24
constexpr long P_K  = 198;  // 24
constexpr long P_V  = 222;  // 24
constexpr long NPLANES = 246;

// packed-weight region (halves); every sub-region is a multiple of 512 halves
constexpr long WBASE = NPLANES * PLANE;      // 16-half aligned
constexpr long W0PK = WBASE;                 // nM=3, nCh=1  -> 1536
constexpr long W1PK = W0PK + 3 * 1 * 512;    // nM=3, nCh=14 -> 21504
constexpr long W2PK = W1PK + 3 * 14 * 512;
constexpr long W3PK = W2PK + 3 * 14 * 512;   // nM=2, nCh=14 -> 14336
constexpr long SKPK = W3PK + 2 * 14 * 512;   // nM=2, nCh=1  -> 1024
constexpr long QPK  = SKPK + 2 * 1 * 512;
constexpr long KPK  = QPK  + 2 * 1 * 512;
constexpr long VPK  = KPK  + 2 * 1 * 512;
constexpr long TOTAL_HALVES = VPK + 2 * 1 * 512;

// ---------------------------------------------------------------------------
// 0) zero the workspace (halos must be 0 every launch; harness poisons d_ws)
// ---------------------------------------------------------------------------
__global__ void fill_zero_u32(uint32_t* __restrict__ p, long n) {
  long i = (long)blockIdx.x * blockDim.x + threadIdx.x;
  if (i < n) p[i] = 0u;
}

// ---------------------------------------------------------------------------
// 1) fp32 input [6,256,256] -> fp16 padded planes (xa = planes 0..2, xb = 3..5)
// ---------------------------------------------------------------------------
__global__ void cvt_input(const float* __restrict__ x, half_t* __restrict__ ws) {
  int i = blockIdx.x * blockDim.x + threadIdx.x;
  if (i >= 6 * IMG_H * IMG_W) return;
  int c = i >> 16;
  int p = i & 65535;
  int y = p >> 8, xx = p & 255;
  ws[(long)c * PLANE + (long)(y + HALO) * WPAD + (xx + HALO)] = (half_t)x[i];
}

// ---------------------------------------------------------------------------
// 2) pack fp32 OIHW weights into the WMMA A-fragment layout:
//    dst[((mt*nChunks + chunk)*32 + lane)*16 + e] = W[m][k]
//    m = mt*16 + (lane&15);  k = chunk*32 + (lane>>4)*8 + (e<8 ? e : e+8)
// ---------------------------------------------------------------------------
__global__ void pack_weights(const float* __restrict__ w, half_t* __restrict__ dst,
                             int Cout, int Cin, int ks, int nChunks, int total) {
  int i = blockIdx.x * blockDim.x + threadIdx.x;
  if (i >= total) return;
  int e    = i & 15;
  int t1   = i >> 4;
  int lane = t1 & 31;
  int t2   = t1 >> 5;
  int chunk = t2 % nChunks;
  int mt    = t2 / nChunks;
  int m  = mt * 16 + (lane & 15);
  int kb = (lane >> 4) * 8;
  int k  = chunk * 32 + kb + (e < 8 ? e : e + 8);
  int KK = Cin * ks * ks;           // OIHW flattens exactly as k = i*ks*ks + r
  float v = (m < Cout && k < KK) ? w[(long)m * KK + k] : 0.0f;
  dst[i] = (half_t)v;
}

// ---------------------------------------------------------------------------
// TDM descriptor build + issue (D# layout per CDNA5 ISA):
//   group0: count=1 | lds_addr | global_addr[56:0] | type=2
//   group1: data_size=2B, tensor dims (t0,t1), tile dims (t0,t1,t2),
//           dim0 stride, dim1 stride
//   group2: tensor_dim2 = t2 (3-D input tile); group3: zero
// ---------------------------------------------------------------------------
#if USE_TDM
__device__ __forceinline__ void tdm_issue(uint32_t lds_off, const void* gptr,
                                          uint32_t t0, uint32_t t1, uint32_t t2,
                                          uint32_t stride0, uint32_t stride1) {
  uint64_t ga = (uint64_t)(size_t)gptr;
  u32x4 g0 = {};
  g0[0] = 1u;                                             // count=1 (valid, user)
  g0[1] = lds_off;                                        // LDS byte address
  g0[2] = (uint32_t)ga;                                   // global_addr[31:0]
  g0[3] = (uint32_t)((ga >> 32) & 0x1FFFFFFu) | (2u << 30); // addr[56:32], type=2
  i32x8 g1 = {};
  g1[0] = (int)(1u << 16);                                // data_size = 1 -> 2B
  g1[1] = (int)((t0 & 0xFFFFu) << 16);                    // tensor_dim0 lo16
  g1[2] = (int)(((t0 >> 16) & 0xFFFFu) | ((t1 & 0xFFFFu) << 16)); // d0 hi | d1 lo
  g1[3] = (int)(((t1 >> 16) & 0xFFFFu) | ((t0 & 0xFFFFu) << 16)); // d1 hi | tile0
  g1[4] = (int)((t1 & 0xFFFFu) | ((t2 & 0xFFFFu) << 16)); // tile1 | tile2
  g1[5] = (int)stride0;                                   // dim0 stride [31:0]
  g1[6] = (int)((stride1 & 0xFFFFu) << 16);               // d0 str hi | d1 str lo16
  g1[7] = (int)(stride1 >> 16);                           // dim1 stride [47:16]
  i32x4 g2 = {};
  g2[0] = (int)t2;                                        // tensor_dim2
  i32x4 g3 = {};
#if defined(__clang_major__) && __clang_major__ >= 23
  i32x8 gz = {};
  __builtin_amdgcn_tensor_load_to_lds(g0, g1, g2, g3, gz, 0);
#else
  __builtin_amdgcn_tensor_load_to_lds(g0, g1, g2, g3, 0);
#endif
}
#endif

// ---------------------------------------------------------------------------
// 3) implicit-GEMM conv (ks==3 pad1, or ks==1) via v_wmma_f32_16x16x32_f16.
//    Block = 4 waves = one 64-pixel row segment. LDS holds the layer's packed
//    weight block + the [Cin][3][66] input tile, both staged by the TDM.
//    Fused: bias, optional ReLU, optional fp16 residual add.
// ---------------------------------------------------------------------------
__global__ __launch_bounds__(128) void conv_wmma(
    const half_t* __restrict__ in, half_t* __restrict__ out,
    const half_t* __restrict__ wpk, const float* __restrict__ bias,
    const half_t* __restrict__ resid,
    int Cin, int Cout, int ks, int nChunks, int relu, int wbytes) {
  extern __shared__ char smem_raw[];
  half_t* lds_w  = (half_t*)smem_raw;
  half_t* lds_in = (half_t*)(smem_raw + wbytes);

  int ty   = blockIdx.x >> 2;                 // output row
  int x0   = (blockIdx.x & 3) << 6;           // first pixel of 64-pixel segment
  int wave = threadIdx.x >> 5;
  int lane = threadIdx.x & 31;
  int lx   = wave * 16 + (lane & 15);         // pixel within segment [0,64)
  int kb   = (lane >> 4) * 8;                 // K sub-offset for this half-wave
  int nM   = (Cout + 15) >> 4;
  int KK   = Cin * ks * ks;
  int wtot = nM * nChunks * 512;              // packed-weight halves

  // ---- stage weights (1-D) and input tile (3-D: 66 x 3 x Cin) into LDS ----
#if USE_TDM
  if (threadIdx.x < 32) {
    uint32_t lw = (uint32_t)(size_t)lds_w;
    uint32_t li = (uint32_t)(size_t)lds_in;
    const half_t* gin = in + (long)(ty + 2) * WPAD + (x0 + 2);  // (-1,-1) corner
    tdm_issue(lw, wpk, (uint32_t)wtot, 1u, 0u, (uint32_t)wtot, 0u);
    tdm_issue(li, gin, 66u, 3u, (uint32_t)Cin, (uint32_t)WPAD, (uint32_t)PLANE);
    __builtin_amdgcn_s_wait_tensorcnt(0);
  }
#else
  for (int idx = threadIdx.x; idx < wtot; idx += 128) lds_w[idx] = wpk[idx];
  int itot = Cin * 3 * 66;
  for (int idx = threadIdx.x; idx < itot; idx += 128) {
    int c = idx / 198, r = idx % 198, yy = r / 66, xx = r % 66;
    lds_in[idx] = in[(long)c * PLANE + (long)(ty + 2 + yy) * WPAD + (x0 + 2 + xx)];
  }
#endif
  __syncthreads();

  v8f acc0 = {}, acc1 = {}, acc2 = {};

#pragma unroll 2
  for (int ch = 0; ch < nChunks; ++ch) {
    // ---- B fragment: im2col gather from the LDS tile -----------------------
    v16h bfrag;
#pragma unroll
    for (int e = 0; e < 16; ++e) {
      int k = ch * 32 + kb + (e < 8 ? e : e + 8);
      half_t v = (half_t)0.0f;
      if (k < KK) {
        int c, dy, dx;
        if (ks == 3) { c = k / 9; int r = k % 9; dy = r / 3; dx = r % 3; }
        else         { c = k; dy = 1; dx = 1; }
        v = lds_in[(c * 3 + dy) * 66 + lx + dx];
      }
      bfrag[e] = v;
    }
    // ---- A fragments: aligned v16h LDS loads, then WMMA --------------------
    {
      v16h a0 = *(const v16h*)(lds_w + (((long)0 * nChunks + ch) * 32 + lane) * 16);
      acc0 = __builtin_amdgcn_wmma_f32_16x16x32_f16(false, a0, false, bfrag,
                                                    (short)0, acc0, false, false);
    }
    if (nM > 1) {
      v16h a1 = *(const v16h*)(lds_w + (((long)1 * nChunks + ch) * 32 + lane) * 16);
      acc1 = __builtin_amdgcn_wmma_f32_16x16x32_f16(false, a1, false, bfrag,
                                                    (short)0, acc1, false, false);
    }
    if (nM > 2) {
      v16h a2 = *(const v16h*)(lds_w + (((long)2 * nChunks + ch) * 32 + lane) * 16);
      acc2 = __builtin_amdgcn_wmma_f32_16x16x32_f16(false, a2, false, bfrag,
                                                    (short)0, acc2, false, false);
    }
  }

  // ---- store: C/D layout -> VGPR r, lane L: M = r + 8*(L>=16), N = L&15 ----
  int mbase = (lane >> 4) * 8;
  for (int mt = 0; mt < 3; ++mt) {
    if (mt >= nM) break;
    v8f acc = (mt == 0) ? acc0 : ((mt == 1) ? acc1 : acc2);
#pragma unroll
    for (int r = 0; r < 8; ++r) {
      int chn = mt * 16 + mbase + r;
      if (chn < Cout) {
        float v = acc[r] + bias[chn];
        if (relu) v = fmaxf(v, 0.0f);
        long off = (long)chn * PLANE + (long)(ty + HALO) * WPAD + (x0 + HALO + lx);
        if (resid) v += (float)resid[off];
        out[off] = (half_t)v;
      }
    }
  }
}

// ---------------------------------------------------------------------------
// 4) 7x7 local attention: per-pixel 49 logits over Q.K, softmax, weighted V.
//    out[0..23] = y + a ;  out[24..47] = b   (fp32, NCHW)
// ---------------------------------------------------------------------------
__global__ __launch_bounds__(256) void attn_kernel(const half_t* __restrict__ ws,
                                                   float* __restrict__ out) {
  int i = blockIdx.x * blockDim.x + threadIdx.x;
  if (i >= IMG_H * IMG_W) return;
  int y = i >> 8, x = i & 255;
  const half_t* qf = ws + P_Q * PLANE;
  const half_t* kf = ws + P_K * PLANE;
  const half_t* vf = ws + P_V * PLANE;
  const half_t* af = ws + P_A * PLANE;
  const half_t* bf = ws + P_B * PLANE;
  long pc = (long)(y + HALO) * WPAD + (x + HALO);

  float qv[24];
#pragma unroll
  for (int c = 0; c < 24; ++c) qv[c] = (float)qf[(long)c * PLANE + pc];

  float logit[49];
  float mx = -1e30f;
  for (int p = 0; p < 49; ++p) {
    int dy = p / 7 - 3, dx = p % 7 - 3;
    long po = (long)(y + HALO + dy) * WPAD + (x + HALO + dx);
    float s = 0.0f;
#pragma unroll
    for (int c = 0; c < 24; ++c) s += qv[c] * (float)kf[(long)c * PLANE + po];
    s *= 0.2041241452319315f;                     // 1/sqrt(24)
    logit[p] = s;
    mx = fmaxf(mx, s);
  }
  float sum = 0.0f;
  for (int p = 0; p < 49; ++p) { float e = __expf(logit[p] - mx); logit[p] = e; sum += e; }
  float inv = 1.0f / sum;

  float yv[24];
#pragma unroll
  for (int c = 0; c < 24; ++c) yv[c] = 0.0f;
  for (int p = 0; p < 49; ++p) {
    int dy = p / 7 - 3, dx = p % 7 - 3;
    long po = (long)(y + HALO + dy) * WPAD + (x + HALO + dx);
    float w = logit[p] * inv;
#pragma unroll
    for (int c = 0; c < 24; ++c) yv[c] += w * (float)vf[(long)c * PLANE + po];
  }
#pragma unroll
  for (int c = 0; c < 24; ++c) {
    out[(long)c * 65536 + i]        = yv[c] + (float)af[(long)c * PLANE + pc];
    out[(long)(24 + c) * 65536 + i] = (float)bf[(long)c * PLANE + pc];
  }
}

// ---------------------------------------------------------------------------
// host-side launch
// ---------------------------------------------------------------------------
extern "C" void kernel_launch(void* const* d_in, const int* in_sizes, int n_in,
                              void* d_out, int out_size, void* d_ws, size_t ws_size,
                              hipStream_t stream) {
  if (ws_size < (size_t)TOTAL_HALVES * 2) return;  // need ~32.4 MB of scratch

  const float* x    = (const float*)d_in[0];
  const float* w0   = (const float*)d_in[1];
  const float* b0   = (const float*)d_in[2];
  const float* w1   = (const float*)d_in[3];
  const float* b1   = (const float*)d_in[4];
  const float* w2   = (const float*)d_in[5];
  const float* b2   = (const float*)d_in[6];
  const float* w3   = (const float*)d_in[7];
  const float* b3   = (const float*)d_in[8];
  const float* skw  = (const float*)d_in[9];
  const float* skb  = (const float*)d_in[10];
  const float* qw   = (const float*)d_in[11];
  const float* qb   = (const float*)d_in[12];
  const float* kw   = (const float*)d_in[13];
  const float* kb_  = (const float*)d_in[14];
  const float* vw   = (const float*)d_in[15];
  const float* vb   = (const float*)d_in[16];
  (void)in_sizes; (void)n_in; (void)out_size;

  half_t* ws = (half_t*)d_ws;
  float*  out = (float*)d_out;

  // 0) zero workspace (re-establish halos every launch)
  {
    long ndw = TOTAL_HALVES / 2;
    int blocks = (int)((ndw + 255) / 256);
    fill_zero_u32<<<blocks, 256, 0, stream>>>((uint32_t*)d_ws, ndw);
  }
  // 1) fp32 -> fp16 padded input planes
  cvt_input<<<(6 * 65536 + 255) / 256, 256, 0, stream>>>(x, ws);

  // 2) pack all weights into WMMA A-fragment layout
  auto pack = [&](const float* w, long dstOff, int Cout, int Cin, int ks, int nCh) {
    int nM = (Cout + 15) >> 4;
    int total = nM * nCh * 32 * 16;
    pack_weights<<<(total + 255) / 256, 256, 0, stream>>>(w, ws + dstOff, Cout, Cin,
                                                          ks, nCh, total);
  };
  pack(w0,  W0PK, 48,  3, 3,  1);
  pack(w1,  W1PK, 48, 48, 3, 14);
  pack(w2,  W2PK, 48, 48, 3, 14);
  pack(w3,  W3PK, 24, 48, 3, 14);
  pack(skw, SKPK, 24,  3, 1,  1);
  pack(qw,  QPK,  24, 24, 1,  1);
  pack(kw,  KPK,  24, 24, 1,  1);
  pack(vw,  VPK,  24, 24, 1,  1);

  // 3) conv launches: 1024 blocks (256 rows x 4 segments), 4 waves per block
  auto conv = [&](long inP, long outP, long wOff, const float* bias, long resP,
                  int Cin, int Cout, int ks, int nCh, int relu) {
    int nM = (Cout + 15) >> 4;
    int wbytes = nM * nCh * 512 * 2;                       // multiple of 32
    int ibytes = ((Cin * 3 * 66 * 2) + 31) & ~31;
    size_t smem = (size_t)wbytes + (size_t)ibytes;
    conv_wmma<<<1024, 128, smem, stream>>>(ws + inP * PLANE, ws + outP * PLANE,
                                           ws + wOff, bias,
                                           resP < 0 ? (const half_t*)nullptr
                                                    : ws + resP * PLANE,
                                           Cin, Cout, ks, nCh, relu, wbytes);
  };

  // submodel(xa) -> a ; submodel(xb) -> b
  const long srcP[2] = {P_XA, P_XB};
  const long dstP[2] = {P_A, P_B};
  for (int s = 0; s < 2; ++s) {
    conv(srcP[s], P_T1, W0PK, b0, -1, 3, 48, 3, 1, 1);
    conv(P_T1,    P_T2, W1PK, b1, -1, 48, 48, 3, 14, 1);
    conv(P_T2,    P_T1, W2PK, b2, -1, 48, 48, 3, 14, 1);
    conv(srcP[s], P_SK, SKPK, skb, -1, 3, 24, 1, 1, 0);          // skip conv
    conv(P_T1, dstP[s], W3PK, b3, P_SK, 48, 24, 3, 14, 1);       // relu then +skip
  }

  // Q/K/V 1x1 convs
  conv(P_B, P_Q, QPK, qb,  -1, 24, 24, 1, 1, 0);
  conv(P_A, P_K, KPK, kb_, -1, 24, 24, 1, 1, 0);
  conv(P_A, P_V, VPK, vb,  -1, 24, 24, 1, 1, 0);

  // 4) local attention + output assembly
  attn_kernel<<<(65536 + 255) / 256, 256, 0, stream>>>(ws, out);
}
